// NARM_49744311222896
// MI455X (gfx1250) — compile-verified, matrix operand order checked by reading, not run on hardware
//
#include <hip/hip_runtime.h>
#include <hip/hip_bf16.h>

// B=16, T=256, D=128, H=128 — NARM: 2 GRUs + causal additive attention.
#define Bn   16
#define Tn   256
#define Dn   128
#define Hn   128
#define NH3  384   // 3*H

typedef __attribute__((ext_vector_type(16))) __bf16 v16bf;
typedef __attribute__((ext_vector_type(8)))  __bf16 v8bf;
typedef __attribute__((ext_vector_type(8)))  float  v8f;

__device__ __forceinline__ float sigmoidf(float x) {
    return 1.0f / (1.0f + __expf(-x));
}

__device__ __forceinline__ v16bf cat8(v8bf lo, v8bf hi) {
    return __builtin_shufflevector(lo, hi, 0, 1, 2, 3, 4, 5, 6, 7,
                                           8, 9, 10, 11, 12, 13, 14, 15);
}

// ---- WMMA fragment layouts (ISA 7.12.2):
// A (16x32 MxK): lane 0-15 row=lane, K {k0..k0+7, k0+16..k0+23};
//                lane 16-31 same rows, K {k0+8..k0+15, k0+24..k0+31}.
// B (32x16 KxN): mirrored, one column per lane.

// A fragment from fp32 row-major source (cvt on the fly; contiguous b128 loads).
__device__ __forceinline__ v16bf load_a_f32(const float* A, int lda, int row0, int k0, int lane) {
    const float* p = A + (size_t)(row0 + (lane & 15)) * lda + k0 + ((lane >> 4) << 3);
    v16bf a;
#pragma unroll
    for (int i = 0; i < 8; ++i) a[i] = (__bf16)p[i];
#pragma unroll
    for (int i = 0; i < 8; ++i) a[8 + i] = (__bf16)p[16 + i];
    return a;
}

// A fragment from bf16 row-major source (pure 16B vector loads, no cvt).
__device__ __forceinline__ v16bf load_a_bf16_rm(const __bf16* A, int lda, int k0, int lane) {
    const __bf16* p = A + (lane & 15) * lda + k0 + ((lane >> 4) << 3);
    return cat8(*(const v8bf*)p, *(const v8bf*)(p + 16));
}

// B fragment from a pre-swizzled fragment buffer: frag f = 32 contiguous bytes/lane.
__device__ __forceinline__ v16bf load_frag(const __bf16* base, int f) {
    const v8bf* p = (const v8bf*)(base + (size_t)f * 16);
    return cat8(p[0], p[1]);
}

// B fragment gathered from fp32 row-major [K, ldb] (used only where B is runtime data).
__device__ __forceinline__ v16bf load_b_f32(const float* B, int ldb, int k0, int col0, int lane) {
    const int c  = col0 + (lane & 15);
    const int kb = k0 + ((lane >> 4) << 3);
    v16bf b;
#pragma unroll
    for (int i = 0; i < 8; ++i) b[i] = (__bf16)B[(size_t)(kb + i) * ldb + c];
#pragma unroll
    for (int i = 0; i < 8; ++i) b[8 + i] = (__bf16)B[(size_t)(kb + 16 + i) * ldb + c];
    return b;
}

// ---------------------------------------------------------------------------
// Kernel 0: swizzle a fp32 [K, ldb] weight matrix into bf16 WMMA B-fragment
// layout: [ntile][k0i][lane][16 bf16].  One fragment (32B) per thread-iter.
__global__ void swizzle_kernel(const float* __restrict__ src, __bf16* __restrict__ dst,
                               int ntiles, int nk, int ldb) {
    const int total = ntiles * nk * 32;
    for (int f = blockIdx.x * blockDim.x + threadIdx.x; f < total; f += gridDim.x * blockDim.x) {
        const int lane = f & 31;
        const int k0i  = (f >> 5) % nk;
        const int nt   = (f >> 5) / nk;
        const int col  = nt * 16 + (lane & 15);
        const int kb   = k0i * 32 + ((lane >> 4) << 3);
        __bf16* d = dst + (size_t)f * 16;
#pragma unroll
        for (int i = 0; i < 8; ++i) d[i]     = (__bf16)src[(size_t)(kb + i) * ldb + col];
#pragma unroll
        for (int i = 0; i < 8; ++i) d[8 + i] = (__bf16)src[(size_t)(kb + 16 + i) * ldb + col];
    }
}

// ---------------------------------------------------------------------------
// Kernel 1: xw = x @ W + b[0] for both GRUs.  M=4096, K=128, N=384.
__global__ void proj_kernel(const float* __restrict__ x,
                            const __bf16* __restrict__ wfg, const __bf16* __restrict__ wfl,
                            const float* __restrict__ bg,   const float* __restrict__ bl,
                            float* __restrict__ xwg, float* __restrict__ xwl) {
    const int lane = threadIdx.x & 31;
    const int wave = threadIdx.x >> 5;
    const int m0 = blockIdx.x * 16;
    const int nt = blockIdx.y * 4 + wave;          // N-tile index (0..23)
    const __bf16* wf  = blockIdx.z ? wfl : wfg;
    const float* bias = blockIdx.z ? bl : bg;      // bias row 0 (input bias)
    float*       xw   = blockIdx.z ? xwl : xwg;

    v8f c = {};
#pragma unroll
    for (int k0i = 0; k0i < 4; ++k0i) {
        v16bf a = load_a_f32(x, Dn, m0, k0i * 32, lane);
        v16bf b = load_frag(wf, (nt * 4 + k0i) * 32 + lane);
        c = __builtin_amdgcn_wmma_f32_16x16x32_bf16(false, a, false, b, (short)0, c, false, false);
    }
    const int col   = nt * 16 + (lane & 15);
    const int rbase = m0 + ((lane >> 4) << 3);
    const float bv  = bias[col];
#pragma unroll
    for (int r = 0; r < 8; ++r)
        xw[(size_t)(rbase + r) * NH3 + col] = c[r] + bv;
}

// ---------------------------------------------------------------------------
// Kernel 2: sequential GRU (reset_after=true). One workgroup per GRU.
// 256 threads = 8 waves; wave w owns H-columns [16w,16w+16) -> z/r/h gate tiles.
// U: pre-swizzled bf16 fragments, async-copied global->LDS (ASYNCcnt path).
// h: fp32 in registers (gate math) + bf16 row-major LDS copy (WMMA A side).
__global__ void gru_kernel(const float* __restrict__ xw_g, const float* __restrict__ xw_l,
                           const __bf16* __restrict__ ufg, const __bf16* __restrict__ ufl,
                           const float* __restrict__ bg,   const float* __restrict__ bl,
                           const unsigned char* __restrict__ mask,
                           float* __restrict__ hg_buf, float* __restrict__ hl_buf,
                           float* __restrict__ out) {
    extern __shared__ __align__(16) __bf16 U_s[];    // 24 tiles * 4 k * 32 lanes * 16 = 96 KB
    __shared__ __align__(16) __bf16 h_bf[Bn * Hn];   // bf16 h for WMMA A (4 KB)
    __shared__ float b1_s[NH3];

    const int g = blockIdx.x;                        // 0 = global GRU, 1 = local GRU
    const float*  xw = g ? xw_l : xw_g;
    const __bf16* uf = g ? ufl : ufg;
    const float*  bb = g ? bl : bg;
    float* hbuf = g ? hl_buf : hg_buf;

    // Async-copy swizzled U into LDS: 24 iterations of 256 lanes x 16B.
    const unsigned lds_base = (unsigned)(unsigned long long)(void*)U_s;
#pragma unroll 4
    for (int it = 0; it < (Hn * NH3 * 2) / (256 * 16); ++it) {
        const int idx = it * 256 + threadIdx.x;
        const unsigned laddr = lds_base + idx * 16;
        const unsigned long long gaddr = (unsigned long long)uf + (unsigned long long)idx * 16;
        asm volatile("global_load_async_to_lds_b128 %0, %1, off"
                     :: "v"(laddr), "v"(gaddr) : "memory");
    }
    for (int i = threadIdx.x; i < NH3;     i += blockDim.x) b1_s[i] = bb[NH3 + i];
    for (int i = threadIdx.x; i < Bn * Hn; i += blockDim.x) h_bf[i] = (__bf16)0.0f;
    asm volatile("s_wait_asynccnt 0x0" ::: "memory");
    __syncthreads();

    const int lane  = threadIdx.x & 31;
    const int wave  = threadIdx.x >> 5;
    const int col16 = lane & 15;
    const int rhalf = (lane >> 4) << 3;
    const int hcol  = wave * 16 + col16;
    const float b1z = b1_s[hcol];
    const float b1r = b1_s[Hn + hcol];
    const float b1h = b1_s[2 * Hn + hcol];

    float hreg[8];                                   // h[rhalf+r][hcol], owned by this thread
#pragma unroll
    for (int r = 0; r < 8; ++r) hreg[r] = 0.0f;

    for (int t = 0; t < Tn; ++t) {
        // inner = h @ U + b[1] : M=16, K=128, three N=16 gate tiles per wave.
        v8f cz = {}, cr = {}, ch = {};
#pragma unroll
        for (int k0i = 0; k0i < 4; ++k0i) {
            v16bf a  = load_a_bf16_rm(h_bf, Hn, k0i * 32, lane);
            v16bf bz = load_frag(U_s, ((wave)      * 4 + k0i) * 32 + lane);
            v16bf br = load_frag(U_s, ((wave + 8)  * 4 + k0i) * 32 + lane);
            v16bf bh = load_frag(U_s, ((wave + 16) * 4 + k0i) * 32 + lane);
            cz = __builtin_amdgcn_wmma_f32_16x16x32_bf16(false, a, false, bz, (short)0, cz, false, false);
            cr = __builtin_amdgcn_wmma_f32_16x16x32_bf16(false, a, false, br, (short)0, cr, false, false);
            ch = __builtin_amdgcn_wmma_f32_16x16x32_bf16(false, a, false, bh, (short)0, ch, false, false);
        }

        float hn[8];
#pragma unroll
        for (int r = 0; r < 8; ++r) {
            const int bidx = rhalf + r;              // batch row
            const float* xrow = xw + (size_t)(bidx * Tn + t) * NH3;
            const float z  = sigmoidf(xrow[hcol]          + cz[r] + b1z);
            const float rg = sigmoidf(xrow[Hn + hcol]     + cr[r] + b1r);
            const float hh = tanhf(xrow[2 * Hn + hcol] + rg * (ch[r] + b1h));
            const float hv = z * hreg[r] + (1.0f - z) * hh;
            hn[r] = mask[bidx * Tn + t] ? hv : hreg[r];
            hreg[r] = hn[r];
        }
        __syncthreads();                             // all A-fragment reads of h done
#pragma unroll
        for (int r = 0; r < 8; ++r) {
            const int bidx = rhalf + r;
            h_bf[bidx * Hn + hcol] = (__bf16)hn[r];
            hbuf[(size_t)(bidx * Tn + t) * Hn + hcol] = hn[r];
            if (g == 0)                              // h_g -> out[..., 0:H]
                out[(size_t)(bidx * Tn + t) * (2 * Hn) + hcol] = hn[r];
        }
        if (t + 1 < Tn)                              // gfx1250 global_prefetch_b8
            __builtin_prefetch(xw + (size_t)(rhalf * Tn + t + 1) * NH3, 0, 0);
        __syncthreads();                             // h_bf update visible before next step
    }
}

// ---------------------------------------------------------------------------
// Kernel 3: q = h_g @ A2, k = h_l @ A1.  M=4096, K=128, N=128.
__global__ void attn_proj_kernel(const float* __restrict__ hg,  const float* __restrict__ hl,
                                 const __bf16* __restrict__ a1f, const __bf16* __restrict__ a2f,
                                 float* __restrict__ qb, float* __restrict__ kb) {
    const int lane = threadIdx.x & 31;
    const int wave = threadIdx.x >> 5;
    const int m0 = blockIdx.x * 16;
    const int nt = blockIdx.y * 4 + wave;            // N-tile (0..7)
    const float*  Hm = blockIdx.z ? hl : hg;
    const __bf16* Af = blockIdx.z ? a1f : a2f;
    float*        O  = blockIdx.z ? kb : qb;

    v8f c = {};
#pragma unroll
    for (int k0i = 0; k0i < 4; ++k0i) {
        v16bf a = load_a_f32(Hm, Hn, m0, k0i * 32, lane);
        v16bf b = load_frag(Af, (nt * 4 + k0i) * 32 + lane);
        c = __builtin_amdgcn_wmma_f32_16x16x32_bf16(false, a, false, b, (short)0, c, false, false);
    }
    const int col   = nt * 16 + (lane & 15);
    const int rbase = m0 + ((lane >> 4) << 3);
#pragma unroll
    for (int r = 0; r < 8; ++r)
        O[(size_t)(rbase + r) * Hn + col] = c[r];
}

// ---------------------------------------------------------------------------
// Kernel 4: scores[b,i,j] = j<=i ? v . sigmoid(q[b,i,:]+k[b,j,:]) : 0
__global__ void scores_kernel(const float* __restrict__ q, const float* __restrict__ k,
                              const float* __restrict__ v, float* __restrict__ scores) {
    const int bi = blockIdx.x;                       // b*T + i
    const int b  = bi >> 8;
    const int i  = bi & (Tn - 1);
    __shared__ __align__(16) float qs[Hn];
    __shared__ __align__(16) float vs[Hn];
    if (threadIdx.x < Hn) {
        qs[threadIdx.x] = q[(size_t)bi * Hn + threadIdx.x];
        vs[threadIdx.x] = v[threadIdx.x];
    }
    __syncthreads();

    const int j = threadIdx.x;
    float s = 0.0f;
    if (j <= i) {
        const float4* k4 = (const float4*)(k + (size_t)(b * Tn + j) * Hn);
        const float4* q4 = (const float4*)qs;
        const float4* v4 = (const float4*)vs;
#pragma unroll 4
        for (int h = 0; h < Hn / 4; ++h) {
            const float4 kv = k4[h], qv = q4[h], vv = v4[h];
            s += vv.x * sigmoidf(qv.x + kv.x);
            s += vv.y * sigmoidf(qv.y + kv.y);
            s += vv.z * sigmoidf(qv.z + kv.z);
            s += vv.w * sigmoidf(qv.w + kv.w);
        }
    }
    scores[(size_t)bi * Tn + j] = s;
}

// ---------------------------------------------------------------------------
// Kernel 5: c_l[b] = scores[b] @ h_l[b] -> out[..., H:2H].  Per b: 256x256x128.
__global__ void cl_kernel(const float* __restrict__ scores, const float* __restrict__ hl,
                          float* __restrict__ out) {
    const int lane = threadIdx.x & 31;
    const int wave = threadIdx.x >> 5;
    const int b  = blockIdx.z;
    const int m0 = blockIdx.x * 16;
    const int n0 = blockIdx.y * 64 + wave * 16;
    const float* S  = scores + (size_t)b * Tn * Tn;
    const float* Hb = hl + (size_t)b * Tn * Hn;

    v8f c = {};
#pragma unroll
    for (int k0 = 0; k0 < Tn; k0 += 32) {
        v16bf a  = load_a_f32(S, Tn, m0, k0, lane);
        v16bf bm = load_b_f32(Hb, Hn, k0, n0, lane);
        c = __builtin_amdgcn_wmma_f32_16x16x32_bf16(false, a, false, bm, (short)0, c, false, false);
    }
    const int col   = n0 + (lane & 15);
    const int rbase = m0 + ((lane >> 4) << 3);
#pragma unroll
    for (int r = 0; r < 8; ++r)
        out[(size_t)(b * Tn + rbase + r) * (2 * Hn) + Hn + col] = c[r];
}

// ---------------------------------------------------------------------------
extern "C" void kernel_launch(void* const* d_in, const int* in_sizes, int n_in,
                              void* d_out, int out_size, void* d_ws, size_t ws_size,
                              hipStream_t stream) {
    const float* x  = (const float*)d_in[0];
    const unsigned char* mask = (const unsigned char*)d_in[1];   // bool[B,T]
    const float* Wg = (const float*)d_in[2];
    const float* Ug = (const float*)d_in[3];
    const float* bg = (const float*)d_in[4];
    const float* Wl = (const float*)d_in[5];
    const float* Ul = (const float*)d_in[6];
    const float* bl = (const float*)d_in[7];
    const float* A1 = (const float*)d_in[8];
    const float* A2 = (const float*)d_in[9];
    const float* v  = (const float*)d_in[10];
    float* out = (float*)d_out;

    float* ws  = (float*)d_ws;
    float* xwg = ws;                         // [4096,384] fp32
    float* xwl = xwg + 4096 * NH3;
    float* hg  = xwl + 4096 * NH3;           // [4096,128] fp32
    float* hl  = hg  + 4096 * Hn;
    float* qb  = hl  + 4096 * Hn;            // h_l_2 = h_g@A2
    float* kb  = qb  + 4096 * Hn;            // h_l_1 = h_l@A1
    float* sc  = kb  + 4096 * Hn;            // [16,256,256] fp32
    float* p   = sc  + (size_t)Bn * Tn * Tn; // bf16 fragment buffers below
    __bf16* wfg = (__bf16*)p;                // 24*4*32*16 bf16 = 24576 floats
    __bf16* wfl = (__bf16*)(p + 24576);
    __bf16* ufg = (__bf16*)(p + 2 * 24576);
    __bf16* ufl = (__bf16*)(p + 3 * 24576);
    __bf16* a1f = (__bf16*)(p + 4 * 24576);  // 8*4*32*16 bf16 = 8192 floats
    __bf16* a2f = (__bf16*)(p + 4 * 24576 + 8192);

    // Pre-swizzle all weights into WMMA B-fragment layout (bf16).
    swizzle_kernel<<<12, 256, 0, stream>>>(Wg, wfg, 24, 4, NH3);
    swizzle_kernel<<<12, 256, 0, stream>>>(Wl, wfl, 24, 4, NH3);
    swizzle_kernel<<<12, 256, 0, stream>>>(Ug, ufg, 24, 4, NH3);
    swizzle_kernel<<<12, 256, 0, stream>>>(Ul, ufl, 24, 4, NH3);
    swizzle_kernel<<<4,  256, 0, stream>>>(A1, a1f, 8, 4, Hn);
    swizzle_kernel<<<4,  256, 0, stream>>>(A2, a2f, 8, 4, Hn);

    proj_kernel<<<dim3(Bn * Tn / 16, NH3 / 64, 2), 128, 0, stream>>>(
        x, wfg, wfl, bg, bl, xwg, xwl);

    const size_t gru_lds = (size_t)Hn * NH3 * 2;     // 96 KB dynamic (bf16 U frags)
    gru_kernel<<<2, 256, gru_lds, stream>>>(
        xwg, xwl, ufg, ufl, bg, bl, mask, hg, hl, out);

    attn_proj_kernel<<<dim3(Bn * Tn / 16, Hn / 64, 2), 128, 0, stream>>>(
        hg, hl, a1f, a2f, qb, kb);

    scores_kernel<<<Bn * Tn, Tn, 0, stream>>>(qb, kb, v, sc);

    cl_kernel<<<dim3(Tn / 16, Hn / 64, Bn), 128, 0, stream>>>(sc, hl, out);
}
